// CachedHeavyRecentAttentionMasker_23536420782113
// MI455X (gfx1250) — compile-verified, matrix operand order checked by reading, not run on hardware
//
#include <hip/hip_runtime.h>

// ---------------- problem constants (from reference) ----------------
#define HEADS 28
#define GS    7
#define NGRP  4           // HEADS / GS
#define L     2048        // q_len == k_len
#define HB    204         // heavy budget = int(0.1*2048)
#define RB    204         // recent budget
#define QCHUNKS 8
#define ROWS_PER_BLOCK (L / QCHUNKS)   // 256
#define SC_ROWS 32        // rows staged per WMMA super-chunk (K of wmma)
#define KCHUNK 512        // k columns staged in LDS at a time
#define PITCH  40         // f16 pitch (32 q cols + pad) -> 80B, 16B aligned
#define FLT_MIN_VAL -3.4028234663852886e+38f

typedef __attribute__((ext_vector_type(16))) _Float16 v16h;
typedef __attribute__((ext_vector_type(8)))  float    v8f;

// ---------------- kernel 0: init workspace ----------------
__global__ __launch_bounds__(256) void hho_init(float* imp, unsigned* heavy, unsigned* cnt) {
    int i = blockIdx.x * 256 + threadIdx.x;
    if (i < HEADS * L) imp[i] = 0.f;
    if (i < NGRP * (L / 32)) heavy[i] = 0u;
    if (i == 0) *cnt = 0u;
}

// ---------------- kernel 1: importance via softmax + WMMA column-sum ----------------
// grid = HEADS*QCHUNKS blocks, 256 threads (8 waves). Each wave owns 4 q-rows per
// 32-row super-chunk; probs are staged transposed (f16) in LDS and reduced over q
// with v_wmma_f32_16x16x32_f16 (A = all-ones => D[m][n] = sum_K B[K][n]).
// Per-superchunk results are folded into a block-level LDS accumulator (each wave
// owns disjoint k-slices -> no atomics), avoiding any persistent VGPR accumulator
// bank (round-2 version spilled 128 VGPRs to scratch).
__global__ __launch_bounds__(256) void hho_importance(const float* __restrict__ attn,
                                                      float* __restrict__ importance) {
    __shared__ _Float16 ldsT[KCHUNK * PITCH];   // 40 KB: [k_local][q_col]
    __shared__ float    ldsImp[L];              //  8 KB: block-level importance

    const int tid  = threadIdx.x;
    const int lane = tid & 31;
    const int wave = tid >> 5;
    const int h  = blockIdx.x / QCHUNKS;
    const int qc = blockIdx.x % QCHUNKS;
    const long rowBase = ((long)h * L + (long)qc * ROWS_PER_BLOCK) * (long)L;

    for (int k = tid; k < L; k += 256) ldsImp[k] = 0.f;   // first c-loop barrier covers this

    v16h ones;
    #pragma unroll
    for (int i = 0; i < 16; ++i) ones[i] = (_Float16)1.0f;

    v8f zacc;
    #pragma unroll
    for (int j = 0; j < 8; ++j) zacc[j] = 0.f;

    float mrow[4], rinv[4];

    for (int sc = 0; sc < ROWS_PER_BLOCK / SC_ROWS; ++sc) {
        // -------- phase A: exact softmax stats for this wave's 4 rows --------
        #pragma unroll
        for (int r = 0; r < 4; ++r) {
            const int qloc = sc * SC_ROWS + wave * 4 + r;
            const float* row = attn + rowBase + (long)qloc * L;
            float4 x[16];
            float m = FLT_MIN_VAL;
            #pragma unroll
            for (int i = 0; i < 16; ++i) x[i] = ((const float4*)row)[i * 32 + lane];
            #pragma unroll
            for (int i = 0; i < 16; ++i)
                m = fmaxf(m, fmaxf(fmaxf(x[i].x, x[i].y), fmaxf(x[i].z, x[i].w)));
            #pragma unroll
            for (int off = 16; off; off >>= 1) m = fmaxf(m, __shfl_xor(m, off, 32));
            float s = 0.f;
            #pragma unroll
            for (int i = 0; i < 16; ++i) {
                s += __expf(x[i].x - m) + __expf(x[i].y - m) +
                     __expf(x[i].z - m) + __expf(x[i].w - m);
            }
            #pragma unroll
            for (int off = 16; off; off >>= 1) s += __shfl_xor(s, off, 32);
            mrow[r] = m;
            rinv[r] = 1.f / s;
        }
        // -------- phase B: per k-chunk: stage probs (L2-hot reload) then WMMA --------
        for (int c = 0; c < 4; ++c) {
            __syncthreads();   // previous chunk's WMMA reads complete / ldsImp init done
            #pragma unroll
            for (int r = 0; r < 4; ++r) {
                const int qloc = sc * SC_ROWS + wave * 4 + r;
                const int qcol = wave * 4 + r;                 // 0..31 inside super-chunk
                const float* row = attn + rowBase + (long)qloc * L + c * KCHUNK;
                float4 xs[4];
                #pragma unroll
                for (int i = 0; i < 4; ++i) xs[i] = ((const float4*)row)[i * 32 + lane];
                #pragma unroll
                for (int i = 0; i < 4; ++i) {
                    const int kl = i * 128 + lane * 4;         // k within chunk
                    ldsT[(kl + 0) * PITCH + qcol] = (_Float16)(__expf(xs[i].x - mrow[r]) * rinv[r]);
                    ldsT[(kl + 1) * PITCH + qcol] = (_Float16)(__expf(xs[i].y - mrow[r]) * rinv[r]);
                    ldsT[(kl + 2) * PITCH + qcol] = (_Float16)(__expf(xs[i].z - mrow[r]) * rinv[r]);
                    ldsT[(kl + 3) * PITCH + qcol] = (_Float16)(__expf(xs[i].w - mrow[r]) * rinv[r]);
                }
            }
            __syncthreads();
            // wave handles 4 tiles of 16 k-columns: B[K=0..31][N=0..15].
            // Preload all 4 B tiles, then issue the 4 WMMAs back-to-back.
            v16h b[4];
            #pragma unroll
            for (int t = 0; t < 4; ++t) {
                const int k0 = (wave * 4 + t) * 16;
                const _Float16* src = &ldsT[(k0 + (lane & 15)) * PITCH + (lane >> 4) * 16];
                #pragma unroll
                for (int e = 0; e < 16; ++e) b[t][e] = src[e];
            }
            v8f d[4];
            #pragma unroll
            for (int t = 0; t < 4; ++t) {
                d[t] = __builtin_amdgcn_wmma_f32_16x16x32_f16(
                    false, ones, false, b[t], (short)0, zacc, false, false);
            }
            // D layout: VGPR0, lanes 0..15 hold column sums for N = lane.
            // Each wave owns disjoint k-slices -> plain LDS RMW, no atomics.
            if (lane < 16) {
                #pragma unroll
                for (int t = 0; t < 4; ++t) {
                    const int k = c * KCHUNK + (wave * 4 + t) * 16 + lane;
                    ldsImp[k] += d[t][0];
                }
            }
        }
    }
    __syncthreads();
    for (int k = tid; k < L; k += 256)
        atomicAdd(&importance[h * L + k], ldsImp[k]);
}

// ---------------- kernel 2: per-head top-204 -> group heavy bitmask ----------------
__global__ __launch_bounds__(256) void hho_topk(const float* __restrict__ importance,
                                                unsigned* __restrict__ heavy) {
    __shared__ float simp[L];
    __shared__ float ssort[L];
    __shared__ unsigned scnt[256];
    const int h = blockIdx.x;
    const int g = h / GS;
    for (int k = threadIdx.x; k < L; k += 256) {
        float v = importance[h * L + k];
        simp[k] = v;
        ssort[k] = v;
    }
    __syncthreads();
    // bitonic sort ascending
    for (int ksz = 2; ksz <= L; ksz <<= 1) {
        for (int j = ksz >> 1; j > 0; j >>= 1) {
            for (int i = threadIdx.x; i < L; i += 256) {
                int ixj = i ^ j;
                if (ixj > i) {
                    float a = ssort[i], b = ssort[ixj];
                    bool up = (i & ksz) == 0;
                    if ((up && a > b) || (!up && a < b)) { ssort[i] = b; ssort[ixj] = a; }
                }
            }
            __syncthreads();
        }
    }
    const float thr = ssort[L - HB];       // 204th largest
    unsigned cnt = 0;
    for (int k = threadIdx.x; k < L; k += 256) cnt += (simp[k] > thr) ? 1u : 0u;
    scnt[threadIdx.x] = cnt;
    __syncthreads();
    if (threadIdx.x == 0) {
        unsigned tot = 0;
        for (int i = 0; i < 256; ++i) tot += scnt[i];
        int need = HB - (int)tot;          // ties resolved by lowest index (top_k order)
        for (int w = 0; w < L / 32; ++w) {
            unsigned bits = 0;
            for (int b = 0; b < 32; ++b) {
                const int k = w * 32 + b;
                const float v = simp[k];
                bool hv = v > thr;
                if (!hv && v == thr && need > 0) { hv = true; --need; }
                if (hv) bits |= (1u << b);
            }
            atomicOr(&heavy[g * (L / 32) + w], bits);
        }
    }
}

// ---------------- kernel 3: write mask + count kept ----------------
__global__ __launch_bounds__(256) void hho_mask(const unsigned* __restrict__ heavy,
                                                float* __restrict__ out,
                                                unsigned* __restrict__ cnt) {
    __shared__ unsigned wsum[8];
    const int b = blockIdx.x;              // b = h*2048 + q
    const int q = b & (L - 1);
    const int g = (b >> 11) / GS;
    const int k = threadIdx.x * 8;
    const unsigned word = heavy[g * (L / 32) + (k >> 5)];
    float vals[8];
    unsigned c = 0;
    #pragma unroll
    for (int j = 0; j < 8; ++j) {
        const int kk = k + j;
        const bool keep = (kk <= q) && ((((word >> (kk & 31)) & 1u) != 0u) || (kk + RB >= q));
        c += keep ? 1u : 0u;
        vals[j] = keep ? 0.f : FLT_MIN_VAL;
    }
    float4* o = (float4*)(out + ((long)b << 11) + k);
    o[0] = make_float4(vals[0], vals[1], vals[2], vals[3]);
    o[1] = make_float4(vals[4], vals[5], vals[6], vals[7]);
    #pragma unroll
    for (int off = 16; off; off >>= 1) c += __shfl_xor((int)c, off, 32);
    if ((threadIdx.x & 31) == 0) wsum[threadIdx.x >> 5] = c;
    __syncthreads();
    if (threadIdx.x == 0) {
        unsigned t = 0;
        #pragma unroll
        for (int i = 0; i < 8; ++i) t += wsum[i];
        atomicAdd(cnt, t);
    }
}

// ---------------- kernel 4: density scalar ----------------
__global__ void hho_density(const unsigned* __restrict__ cnt, float* __restrict__ out) {
    if (threadIdx.x == 0 && blockIdx.x == 0) {
        double d = (double)(*cnt) / (double)HEADS / ((double)L * (double)(L + 1) * 0.5);
        out[(long)HEADS * L * L] = (float)d;
    }
}

// ---------------- launcher ----------------
extern "C" void kernel_launch(void* const* d_in, const int* in_sizes, int n_in,
                              void* d_out, int out_size, void* d_ws, size_t ws_size,
                              hipStream_t stream) {
    const float* attn = (const float*)d_in[0];   // [1,28,2048,2048] fp32
    float* out = (float*)d_out;                  // mask (28*2048*2048) + density (1)

    float*    imp   = (float*)d_ws;                                   // 28*2048 f32
    unsigned* heavy = (unsigned*)((char*)d_ws + HEADS * L * sizeof(float));  // 4*64 u32
    unsigned* cnt   = heavy + NGRP * (L / 32);                        // 1 u32

    hho_init<<<(HEADS * L + 255) / 256, 256, 0, stream>>>(imp, heavy, cnt);
    hho_importance<<<HEADS * QCHUNKS, 256, 0, stream>>>(attn, imp);
    hho_topk<<<HEADS, 256, 0, stream>>>(imp, heavy);
    hho_mask<<<HEADS * L, 256, 0, stream>>>(heavy, out, cnt);
    hho_density<<<1, 1, 0, stream>>>(cnt, out);
}
